// GCNBasic_33552284516575
// MI455X (gfx1250) — matrix operand order repaired and checked by reading, not compiled, not run on hardware
//
#include <hip/hip_runtime.h>
#include <stdint.h>
#include <stddef.h>

// ---------------------------------------------------------------------------
// GCN forward on MI455X (gfx1250): bf16 WMMA for all GEMMs, fp32 LayerNorm.
// Workspace requirement: ~336 MiB (dominated by bf16 copy of A_hat, 256 MiB).
//
// GEMM wave tile: 32 rows x 64 cols = 8 x v_wmma_f32_16x16x32_bf16 per K-step
// fed by 12 global_load_b128 (~21 flop/byte from cache -> compute bound).
// ---------------------------------------------------------------------------

#define NB   32
#define NN   2048
#define FF   128
#define HH   256
#define KOUT 64
#define LN_EPS 1e-5f

typedef __attribute__((ext_vector_type(16))) __bf16        v16bf;
typedef __attribute__((ext_vector_type(8)))  float         v8f;
typedef __attribute__((ext_vector_type(4)))  unsigned int  v4u;

union BF16Frag { v16bf v; v4u q[2]; unsigned short s[16]; };

__device__ __forceinline__ unsigned short f2bf(float f) {
  union { float f; unsigned int u; } x; x.f = f;
  unsigned int u = x.u;
  u += 0x7fffu + ((u >> 16) & 1u);           // round-to-nearest-even
  return (unsigned short)(u >> 16);
}

#define WMMA_BF16(A, Bm, Cacc) \
  __builtin_amdgcn_wmma_f32_16x16x32_bf16(false, (A), false, (Bm), (short)0, (Cacc), false, false)

// ---------------------------------------------------------------------------
// Cast fp32 -> bf16, vectorized (n must be a multiple of 4).
// ---------------------------------------------------------------------------
__global__ void cast_f32_to_bf16(const float* __restrict__ src,
                                 unsigned short* __restrict__ dst, long n4) {
  long i      = (long)blockIdx.x * blockDim.x + threadIdx.x;
  long stride = (long)gridDim.x * blockDim.x;
  const float4* s4 = (const float4*)src;
  for (; i < n4; i += stride) {
    float4 v = s4[i];
    unsigned long long p = (unsigned long long)f2bf(v.x)
                         | ((unsigned long long)f2bf(v.y) << 16)
                         | ((unsigned long long)f2bf(v.z) << 32)
                         | ((unsigned long long)f2bf(v.w) << 48);
    ((unsigned long long*)dst)[i] = p;
  }
}

// ---------------------------------------------------------------------------
// Transpose + cast: src (nb, R, C) fp32 -> dst (nb, C, R) bf16.
// ---------------------------------------------------------------------------
__global__ void tcast_bf16(const float* __restrict__ src,
                           unsigned short* __restrict__ dst,
                           int R, int C, int nb) {
  long per    = (long)R * C;
  long total  = (long)nb * per;
  long i      = (long)blockIdx.x * blockDim.x + threadIdx.x;
  long stride = (long)gridDim.x * blockDim.x;
  for (; i < total; i += stride) {
    long bb  = i / per;
    long rem = i - bb * per;
    long c   = rem / R;
    long r   = rem - c * R;
    dst[i] = f2bf(src[bb * per + r * C + c]);
  }
}

__global__ void zero_f32(float* __restrict__ p, int n) {
  int i = blockIdx.x * blockDim.x + threadIdx.x;
  if (i < n) p[i] = 0.0f;
}

// ---------------------------------------------------------------------------
// Big batched GEMM: C[b] = A[b] (MxK) x B[b] (KxNc), all bf16, B given
// transposed as BT (Nc x K) so every WMMA B-fragment is 2 contiguous b128
// loads.  Block tile: 256 rows x 64 cols, 8 waves; wave = 32 rows x 64 cols
// (2 A-frags x 4 B-frags -> 8 accumulators).  K step 32.
// ---------------------------------------------------------------------------
__global__ __launch_bounds__(256)
void wmma_gemm_bf16(const unsigned short* __restrict__ A,
                    const unsigned short* __restrict__ BT,
                    unsigned short* __restrict__ C,
                    int M, int Nc, int Kd) {
  const int b       = blockIdx.z;
  const int rowBase = blockIdx.y * 256;
  const int colBase = blockIdx.x * 64;
  const int tid     = threadIdx.x;
  const int wave    = tid >> 5;
  const int lane    = tid & 31;
  const int lhalf   = lane >> 4;   // 0 | 1
  const int l16     = lane & 15;

  const unsigned short* Ab = A  + (size_t)b * M * Kd;
  const unsigned short* Bb = BT + (size_t)b * Nc * Kd;
  unsigned short*       Cb = C  + (size_t)b * M * Nc;

  // A frag: lane row = l16, k chunk [kb, kb+16), kb=lhalf*16
  const unsigned short* aptr0 = Ab + (size_t)(rowBase + wave * 32 + l16) * Kd + lhalf * 16;
  const unsigned short* aptr1 = aptr0 + (size_t)16 * Kd;
  // B frag: lane col = l16, k chunk [kb2, kb2+16), kb2=lhalf*16
  const unsigned short* bptr  = Bb + (size_t)(colBase + l16) * Kd + lhalf * 16;
  const size_t bstep = (size_t)16 * Kd;

  v8f accA[4] = {};   // rows [wave*32, +16)
  v8f accB[4] = {};   // rows [wave*32+16, +32)

  for (int k0 = 0; k0 < Kd; k0 += 32) {
    BF16Frag af0, af1, bfr[4];
    af0.q[0] = *(const v4u*)(aptr0 + k0);
    af0.q[1] = *(const v4u*)(aptr0 + k0 + 8);
    af1.q[0] = *(const v4u*)(aptr1 + k0);
    af1.q[1] = *(const v4u*)(aptr1 + k0 + 8);
    #pragma unroll
    for (int ct = 0; ct < 4; ++ct) {
      bfr[ct].q[0] = *(const v4u*)(bptr + (size_t)ct * bstep + k0);
      bfr[ct].q[1] = *(const v4u*)(bptr + (size_t)ct * bstep + k0 + 8);
    }
    #pragma unroll
    for (int ct = 0; ct < 4; ++ct) {
      accA[ct] = WMMA_BF16(af0.v, bfr[ct].v, accA[ct]);
      accB[ct] = WMMA_BF16(af1.v, bfr[ct].v, accB[ct]);
    }
  }

  // D layout: vgpr r, lane -> row = r + 8*lhalf, col = l16 (+16*ct)
  #pragma unroll
  for (int ct = 0; ct < 4; ++ct) {
    const int col = colBase + ct * 16 + l16;
    const int crowA = rowBase + wave * 32 + lhalf * 8;
    #pragma unroll
    for (int r = 0; r < 8; ++r)
      Cb[(size_t)(crowA + r) * Nc + col] = f2bf(accA[ct][r]);
    const int crowB = crowA + 16;
    #pragma unroll
    for (int r = 0; r < 8; ++r)
      Cb[(size_t)(crowB + r) * Nc + col] = f2bf(accB[ct][r]);
  }
}

// ---------------------------------------------------------------------------
// Fused: Out = ReLU(LayerNorm(Tin @ W + bias)) for one 16-row x 256-col tile.
// Tin: (NB, M, Kd) bf16 row-major; WT: (HH, Kd) bf16 (W transposed).
// writeOut=1: store bf16 transposed (NB, HH, M) -> feeds next GEMM's BT.
// writeOut=0: accumulate per-column sums into gsum (NB, HH) for mean readout.
// ---------------------------------------------------------------------------
__global__ __launch_bounds__(256)
void wmma_linear_ln(const unsigned short* __restrict__ Tin,
                    const unsigned short* __restrict__ WT,
                    const float* __restrict__ bias,
                    const float* __restrict__ gamma,
                    const float* __restrict__ beta,
                    unsigned short* __restrict__ OutT,
                    float* __restrict__ gsum,
                    int M, int Kd, int writeOut) {
  __shared__ float smemC[16][HH + 8];
  __shared__ float sMu[16];
  __shared__ float sRs[16];

  const int b       = blockIdx.y;
  const int rowBase = blockIdx.x * 16;
  const int tid     = threadIdx.x;
  const int wave    = tid >> 5;
  const int lane    = tid & 31;
  const int lhalf   = lane >> 4;
  const int l16     = lane & 15;

  // wave w covers columns [32w, 32w+32): two 16x16 accumulators
  const unsigned short* aptr = Tin + (size_t)b * M * Kd
                                   + (size_t)(rowBase + l16) * Kd + lhalf * 16;
  const unsigned short* bptr = WT + (size_t)(wave * 32 + l16) * Kd + lhalf * 16;
  const size_t bstep = (size_t)16 * Kd;

  v8f acc0 = {}, acc1 = {};
  for (int k0 = 0; k0 < Kd; k0 += 32) {
    BF16Frag af, b0, b1;
    af.q[0] = *(const v4u*)(aptr + k0);
    af.q[1] = *(const v4u*)(aptr + k0 + 8);
    b0.q[0] = *(const v4u*)(bptr + k0);
    b0.q[1] = *(const v4u*)(bptr + k0 + 8);
    b1.q[0] = *(const v4u*)(bptr + bstep + k0);
    b1.q[1] = *(const v4u*)(bptr + bstep + k0 + 8);
    acc0 = WMMA_BF16(af.v, b0.v, acc0);
    acc1 = WMMA_BF16(af.v, b1.v, acc1);
  }

  // dump accumulators (+bias) into LDS fp32 tile
  {
    const int m0 = lhalf * 8;
    const int n0 = wave * 32 + l16;
    const float bv0 = bias[n0], bv1 = bias[n0 + 16];
    #pragma unroll
    for (int r = 0; r < 8; ++r) {
      smemC[m0 + r][n0]      = acc0[r] + bv0;
      smemC[m0 + r][n0 + 16] = acc1[r] + bv1;
    }
  }
  __syncthreads();

  // per-row mean/var: 16 threads per row, shuffle reduce within the 16-group
  {
    const int rr = tid >> 4, j = tid & 15;
    float s = 0.0f, s2 = 0.0f;
    #pragma unroll
    for (int c = 0; c < 16; ++c) {
      float v = smemC[rr][j * 16 + c];
      s += v; s2 += v * v;
    }
    #pragma unroll
    for (int off = 8; off >= 1; off >>= 1) {
      s  += __shfl_xor(s,  off, 32);
      s2 += __shfl_xor(s2, off, 32);
    }
    if (j == 0) {
      float mu  = s  * (1.0f / HH);
      float var = s2 * (1.0f / HH) - mu * mu;
      sMu[rr] = mu;
      sRs[rr] = rsqrtf(var + LN_EPS);
    }
  }
  __syncthreads();

  // normalize + ReLU; thread t owns column t over all 16 rows
  {
    const int c = tid;
    const float gv = gamma[c], bv = beta[c];
    if (writeOut) {
      union { v4u q[2]; unsigned short s[16]; } o;
      #pragma unroll
      for (int r = 0; r < 16; ++r) {
        float y = (smemC[r][c] - sMu[r]) * sRs[r] * gv + bv;
        o.s[r] = f2bf(fmaxf(y, 0.0f));
      }
      unsigned short* dst = OutT + (size_t)b * HH * M + (size_t)c * M + rowBase;
      *(v4u*)(dst)     = o.q[0];
      *(v4u*)(dst + 8) = o.q[1];
    } else {
      float sum = 0.0f;
      #pragma unroll
      for (int r = 0; r < 16; ++r) {
        float y = (smemC[r][c] - sMu[r]) * sRs[r] * gv + bv;
        sum += fmaxf(y, 0.0f);
      }
      atomicAdd(&gsum[b * HH + c], sum);
    }
  }
}

// ---------------------------------------------------------------------------
// Readout: g = gsum / N ; pred = g@Wa + ba ; logits = g@Wl + bl.
// ---------------------------------------------------------------------------
__global__ void readout_kernel(const float* __restrict__ gsum,
                               const float* __restrict__ Wa, const float* __restrict__ ba,
                               const float* __restrict__ Wl, const float* __restrict__ bl,
                               float* __restrict__ out) {
  const int b = blockIdx.x;
  const int k = threadIdx.x;   // 64 threads
  float pa = 0.0f, pl = 0.0f;
  for (int h = 0; h < HH; ++h) {
    float g = gsum[b * HH + h] * (1.0f / NN);
    pa += g * Wa[h * KOUT + k];
    pl += g * Wl[h * KOUT + k];
  }
  out[b * KOUT + k]             = pa + ba[k];
  out[NB * KOUT + b * KOUT + k] = pl + bl[k];
}

// ---------------------------------------------------------------------------
extern "C" void kernel_launch(void* const* d_in, const int* in_sizes, int n_in,
                              void* d_out, int out_size, void* d_ws, size_t ws_size,
                              hipStream_t stream) {
  (void)in_sizes; (void)n_in; (void)out_size; (void)ws_size;

  const float* A_hat = (const float*)d_in[0];
  const float* X     = (const float*)d_in[1];
  const float* W1    = (const float*)d_in[2];
  const float* b1    = (const float*)d_in[3];
  const float* g1    = (const float*)d_in[4];
  const float* beta1 = (const float*)d_in[5];
  const float* W2    = (const float*)d_in[6];
  const float* b2    = (const float*)d_in[7];
  const float* g2    = (const float*)d_in[8];
  const float* beta2 = (const float*)d_in[9];
  const float* Wa    = (const float*)d_in[10];
  const float* ba    = (const float*)d_in[11];
  const float* Wl    = (const float*)d_in[12];
  const float* bl    = (const float*)d_in[13];
  float* out = (float*)d_out;

  // --- workspace carve-up (256-byte aligned) ---
  uint8_t* p = (uint8_t*)d_ws;
  auto carve = [&](size_t bytes) -> uint8_t* {
    uint8_t* r = p;
    p += (bytes + 255) & ~(size_t)255;
    return r;
  };
  const size_t szA   = (size_t)NB * NN * NN;   // bf16 A_hat
  const size_t szXT  = (size_t)NB * FF * NN;   // bf16 X^T
  const size_t szW1T = (size_t)HH * FF;
  const size_t szW2T = (size_t)HH * HH;
  const size_t szH1T = (size_t)NB * HH * NN;   // bf16 H1^T
  const size_t szTS  = (size_t)NB * NN * HH;   // shared buffer for T and S

  unsigned short* A_bf = (unsigned short*)carve(szA * 2);
  unsigned short* XT   = (unsigned short*)carve(szXT * 2);
  unsigned short* W1T  = (unsigned short*)carve(szW1T * 2);
  unsigned short* W2T  = (unsigned short*)carve(szW2T * 2);
  unsigned short* H1T  = (unsigned short*)carve(szH1T * 2);
  unsigned short* TS   = (unsigned short*)carve(szTS * 2);
  float*          gsum = (float*)carve((size_t)NB * HH * 4);

  // 1) precision conversion / layout transforms
  {
    long n4 = (long)(szA / 4);
    cast_f32_to_bf16<<<(unsigned)(n4 / 256), 256, 0, stream>>>(A_hat, A_bf, n4);
  }
  tcast_bf16<<<4096, 256, 0, stream>>>(X,  XT,  NN, FF, NB);   // (B,N,F)->(B,F,N)
  tcast_bf16<<<128, 256, 0, stream>>>(W1, W1T, FF, HH, 1);     // (F,H)->(H,F)
  tcast_bf16<<<256, 256, 0, stream>>>(W2, W2T, HH, HH, 1);     // (H,H)->(H,H)^T
  zero_f32<<<(NB * HH + 255) / 256, 256, 0, stream>>>(gsum, NB * HH);

  // 2) Layer 1: T = A @ X    (M=2048, Nc=128, K=2048)
  wmma_gemm_bf16<<<dim3(FF / 64, NN / 256, NB), 256, 0, stream>>>(
      A_bf, XT, TS, NN, FF, NN);

  // 3) H1^T = relu(LN(T @ W1 + b1))^T
  wmma_linear_ln<<<dim3(NN / 16, NB), 256, 0, stream>>>(
      TS, W1T, b1, g1, beta1, H1T, nullptr, NN, FF, 1);

  // 4) Layer 2: S = A @ H1   (M=2048, Nc=256, K=2048)
  wmma_gemm_bf16<<<dim3(HH / 64, NN / 256, NB), 256, 0, stream>>>(
      A_bf, H1T, TS, NN, HH, NN);

  // 5) relu(LN(S @ W2 + b2)) -> column sums for mean readout
  wmma_linear_ln<<<dim3(NN / 16, NB), 256, 0, stream>>>(
      TS, W2T, b2, g2, beta2, nullptr, gsum, NN, HH, 0);

  // 6) readout heads
  readout_kernel<<<NB, KOUT, 0, stream>>>(gsum, Wa, ba, Wl, bl, out);
}
